// SEModule_82746839925439
// MI455X (gfx1250) — compile-verified, matrix operand order checked by reading, not run on hardware
//
#include <hip/hip_runtime.h>

// ---------------------------------------------------------------------------
// SE module (int8 fake-quant) for MI455X / gfx1250.
//
// Bandwidth plan (the kernel is memory-bound): 3 reads + 1 write of the
// 205 MB activation tensor = ~822 MB => ~35us at 23.3 TB/s. The two tiny
// GEMMs (64x256x1024, 64x1024x256, ~67 MFLOP) run as signed int8 WMMA
// (v_wmma_i32_16x16x64_iu8) in a single workgroup, exploiting
//   fake_quant(a) @ fake_quant(W).T == s_a*s_W * (q_a_i8 @ q_W_i8.T)
// which is exact (quantized values are integers in [-127,127]).
// ---------------------------------------------------------------------------

typedef __attribute__((ext_vector_type(8))) int v8i;

__device__ __forceinline__ float scale_from_bits(unsigned bits) {
    // scale = max(absmax/127, 1e-8)  (absmax stored as non-negative float bits)
    return fmaxf(__uint_as_float(bits) * (1.0f / 127.0f), 1e-8f);
}

__device__ __forceinline__ int quant1(float v, float inv_s) {
    // round-to-nearest-even (matches jnp.round), clip to [-127,127]
    int q = __float2int_rn(v * inv_s);
    return max(-127, min(127, q));
}

// Build one 16x64 int8 operand fragment from a row-major fp32 tile
// (row r at base + r*stride), quantizing on the fly.
// Layout per ISA 7.12.2 (8-bit A-matrix 16x64): lanes 0-15 hold rows 0-15
// with K-chunks {0-3,4-7,16-19,20-23,+32...} in VGPRs 0..7; lanes 16-31 hold
// the same rows shifted by K+8. B (64x16) uses the mirrored layout (role of
// row index -> N); since our B source W[n,k] is row-major over k, the same
// builder applies with base = &W[n0*stride].
__device__ __forceinline__ v8i build_frag(const float* base, int stride,
                                          int lane, float inv_s) {
    union { v8i v; unsigned u[8]; } out;
    const int r = lane & 15;
    const int khalf = (lane >> 4) << 3;          // 0 or 8
    const float* rowp = base + r * stride + khalf;
#pragma unroll
    for (int v = 0; v < 8; ++v) {
        const int k = ((v >> 1) << 4) + ((v & 1) << 2); // 0,4,16,20,32,36,48,52
        const float* p = rowp + k;
        unsigned b0 = (unsigned)(quant1(p[0], inv_s) & 255);
        unsigned b1 = (unsigned)(quant1(p[1], inv_s) & 255);
        unsigned b2 = (unsigned)(quant1(p[2], inv_s) & 255);
        unsigned b3 = (unsigned)(quant1(p[3], inv_s) & 255);
        out.u[v] = b0 | (b1 << 8) | (b2 << 16) | (b3 << 24);
    }
    return out.v;
}

// ---------------------------------------------------------------------------
__global__ void k_init(unsigned* slots) {
    if (threadIdx.x < 4) slots[threadIdx.x] = 0u;
}

// Grid-stride float4 absmax -> atomicMax on float-bits (order independent).
__global__ void __launch_bounds__(256) k_absmax4(const float4* __restrict__ src,
                                                 int n4, unsigned* slot) {
    __shared__ unsigned smax;
    if (threadIdx.x == 0) smax = 0u;
    __syncthreads();
    float m = 0.0f;
    for (int i = blockIdx.x * blockDim.x + threadIdx.x; i < n4;
         i += gridDim.x * blockDim.x) {
        float4 v = src[i];
        m = fmaxf(m, fmaxf(fmaxf(fabsf(v.x), fabsf(v.y)),
                           fmaxf(fabsf(v.z), fabsf(v.w))));
    }
    atomicMax(&smax, __float_as_uint(m));
    __syncthreads();
    if (threadIdx.x == 0) atomicMax(slot, smax);
}

// Quantized global average pool: one wave32 per (b,c) pair.
// se[b,c] = s_x * sum(clip(round(x/s_x))) / 784 ; also absmax(se) -> slot_se.
__global__ void __launch_bounds__(256) k_pool(const float* __restrict__ x,
                                              float* __restrict__ se,
                                              unsigned* slot_se,
                                              const unsigned* __restrict__ slots) {
    __shared__ unsigned smax;
    if (threadIdx.x == 0) smax = 0u;
    __syncthreads();

    const float s1 = scale_from_bits(slots[0]);
    const float inv_s1 = 1.0f / s1;
    const int wave = threadIdx.x >> 5;
    const int lane = threadIdx.x & 31;
    const int pair = blockIdx.x * (blockDim.x >> 5) + wave;   // 0..65535

    const float* p = x + (size_t)pair * 784;
    int sum = 0;
    for (int i = lane; i < 784; i += 32) sum += quant1(p[i], inv_s1);
#pragma unroll
    for (int off = 16; off > 0; off >>= 1) sum += __shfl_xor(sum, off, 32);

    const float mean = s1 * (float)sum * (1.0f / 784.0f);
    if (lane == 0) {
        se[pair] = mean;
        atomicMax(&smax, __float_as_uint(fabsf(mean)));
    }
    __syncthreads();
    if (threadIdx.x == 0) atomicMax(slot_se, smax);
}

// Single-workgroup MLP: both int8 GEMMs on the WMMA path; LDS-staged hidden
// layer; workgroup barrier supplies the activation scale between the GEMMs.
// Note: fake_quant is idempotent (max element maps to exactly 127*scale), so
// re-quantizing the fq'd pool output yields the same int8 values.
__global__ void __launch_bounds__(512) k_mlp(const float* __restrict__ se,
                                             const float* __restrict__ W1,
                                             const float* __restrict__ b1,
                                             const float* __restrict__ W2,
                                             const float* __restrict__ b2,
                                             float* __restrict__ gate,
                                             const unsigned* __restrict__ slots) {
    __shared__ float hbuf[64 * 256];      // 64 KB hidden activations
    __shared__ unsigned hmax;
    if (threadIdx.x == 0) hmax = 0u;
    __syncthreads();

    const int lane = threadIdx.x & 31;
    const int wave = threadIdx.x >> 5;    // 0..15

    const float s_se = scale_from_bits(slots[1]);
    const float s_w1 = scale_from_bits(slots[2]);
    const float s_w2 = scale_from_bits(slots[3]);
    const float inv_se = 1.0f / s_se;
    const float inv_w1 = 1.0f / s_w1;
    const float inv_w2 = 1.0f / s_w2;

    // ---- GEMM1: h[64,256] = relu(q(se)@q(W1)^T * s_se*s_w1 + b1) ----------
    const float deq1 = s_se * s_w1;
    float lmax = 0.0f;
    for (int t = 0; t < 4; ++t) {
        const int tile = wave * 4 + t;            // 64 tiles of 16x16
        const int m0 = (tile & 3) << 4;           // M = 64
        const int n0 = (tile >> 2) << 4;          // N = 256
        v8i acc = {0, 0, 0, 0, 0, 0, 0, 0};
        for (int kk = 0; kk < 1024; kk += 64) {   // K = 1024
            v8i a = build_frag(se + m0 * 1024 + kk, 1024, lane, inv_se);
            v8i b = build_frag(W1 + n0 * 1024 + kk, 1024, lane, inv_w1);
            acc = __builtin_amdgcn_wmma_i32_16x16x64_iu8(true, a, true, b,
                                                         acc, false, false);
        }
        const int nn = n0 + (lane & 15);
        const int mbase = m0 + ((lane >> 4) << 3);
        const float bias = b1[nn];
#pragma unroll
        for (int r = 0; r < 8; ++r) {             // C/D layout: VGPR r, lane
            float v = (float)acc[r] * deq1 + bias;
            v = fmaxf(v, 0.0f);
            hbuf[(mbase + r) * 256 + nn] = v;
            lmax = fmaxf(lmax, v);                // relu output: abs == value
        }
    }
    atomicMax(&hmax, __float_as_uint(lmax));
    __syncthreads();                              // h + its absmax visible

    const float s_h = fmaxf(__uint_as_float(hmax) * (1.0f / 127.0f), 1e-8f);
    const float inv_h = 1.0f / s_h;
    const float deq2 = s_h * s_w2;

    // ---- GEMM2: gate[64,1024] = hardsigmoid(q(h)@q(W2)^T * s_h*s_w2 + b2) -
    for (int t = 0; t < 16; ++t) {
        const int tile = t * 16 + wave;           // 256 tiles of 16x16
        const int m0 = (tile & 3) << 4;           // M = 64
        const int n0 = (tile >> 2) << 4;          // N = 1024
        v8i acc = {0, 0, 0, 0, 0, 0, 0, 0};
#pragma unroll
        for (int kk = 0; kk < 256; kk += 64) {    // K = 256
            v8i a = build_frag(hbuf + m0 * 256 + kk, 256, lane, inv_h);
            v8i b = build_frag(W2 + n0 * 256 + kk, 256, lane, inv_w2);
            acc = __builtin_amdgcn_wmma_i32_16x16x64_iu8(true, a, true, b,
                                                         acc, false, false);
        }
        const int nn = n0 + (lane & 15);
        const int mbase = m0 + ((lane >> 4) << 3);
        const float bias = b2[nn];
#pragma unroll
        for (int r = 0; r < 8; ++r) {
            float v = (float)acc[r] * deq2 + bias;
            v = __saturatef(v * (1.0f / 6.0f) + 0.5f);   // hardsigmoid
            gate[(mbase + r) * 1024 + nn] = v;
        }
    }
}

// out = x * gate[b,c], float4-vectorized (784/4 = 196 float4s per channel,
// so channel boundaries are float4-aligned).
__global__ void __launch_bounds__(256) k_scale(const float4* __restrict__ x4,
                                               const float* __restrict__ gate,
                                               float4* __restrict__ o4, int n4) {
    int i = blockIdx.x * blockDim.x + threadIdx.x;
    const int stride = gridDim.x * blockDim.x;
    for (; i < n4; i += stride) {
        const float g = gate[i / 196];
        float4 v = x4[i];
        v.x *= g; v.y *= g; v.z *= g; v.w *= g;
        o4[i] = v;
    }
}

// ---------------------------------------------------------------------------
extern "C" void kernel_launch(void* const* d_in, const int* in_sizes, int n_in,
                              void* d_out, int out_size, void* d_ws, size_t ws_size,
                              hipStream_t stream) {
    const float* x  = (const float*)d_in[0];   // [64,1024,28,28]
    const float* W1 = (const float*)d_in[1];   // [256,1024]
    const float* b1 = (const float*)d_in[2];   // [256]
    const float* W2 = (const float*)d_in[3];   // [1024,256]
    const float* b2 = (const float*)d_in[4];   // [1024]
    float* out = (float*)d_out;

    // ws layout: [0..3] absmax slots (x, se, W1, W2) | se[64*1024] | gate[64*1024]
    unsigned* slots = (unsigned*)d_ws;
    float* se   = (float*)d_ws + 16;
    float* gate = (float*)d_ws + 16 + 64 * 1024;

    const int n4x = (64 * 1024 * 28 * 28) / 4;   // 12,845,056 float4s

    k_init<<<1, 32, 0, stream>>>(slots);
    k_absmax4<<<2048, 256, 0, stream>>>((const float4*)x,  n4x,              slots + 0);
    k_absmax4<<<256,  256, 0, stream>>>((const float4*)W1, (256 * 1024) / 4, slots + 2);
    k_absmax4<<<256,  256, 0, stream>>>((const float4*)W2, (1024 * 256) / 4, slots + 3);
    k_pool<<<65536 / 8, 256, 0, stream>>>(x, se, slots + 1, slots);
    k_mlp<<<1, 512, 0, stream>>>(se, W1, b1, W2, b2, gate, slots);
    k_scale<<<4096, 256, 0, stream>>>((const float4*)x, gate, (float4*)out, n4x);
}
